// Memory_72945724555740
// MI455X (gfx1250) — compile-verified, matrix operand order checked by reading, not run on hardware
//
#include <hip/hip_runtime.h>
#include <hip/hip_bf16.h>

// ---------------------------------------------------------------------------
// Problem constants (from reference): B=8, C=256, H=W=64, M=512
// ---------------------------------------------------------------------------
#define BN   8
#define CN   256
#define HWN  4096      // 64*64
#define MN   512
#define WN   64

typedef __attribute__((ext_vector_type(8)))  __bf16 bf16x8;
typedef __attribute__((ext_vector_type(16))) __bf16 bf16x16;
typedef __attribute__((ext_vector_type(8)))  float  v8f;
typedef __attribute__((ext_vector_type(4)))  unsigned int u32x4;
typedef __attribute__((ext_vector_type(4)))  int  i32x4;
typedef __attribute__((ext_vector_type(8)))  int  i32x8;

union Frag {
    bf16x16 f;
    bf16x8  h[2];
};

#if defined(__has_builtin)
#if __has_builtin(__builtin_amdgcn_tensor_load_to_lds) && \
    __has_builtin(__builtin_amdgcn_s_wait_tensorcnt)
#define USE_TDM 1
#endif
#endif

// Scheduler fence: keep the batched operand loads ahead of the WMMA chain so
// the WMMAs run with partial s_wait_dscnt/loadcnt instead of full stalls.
static __device__ __forceinline__ void sched_fence() {
#if defined(__has_builtin)
#if __has_builtin(__builtin_amdgcn_sched_barrier)
    __builtin_amdgcn_sched_barrier(0);
#endif
#endif
}

static __device__ __forceinline__ unsigned short f2bf(float x) {
    unsigned u = __float_as_uint(x);
    u += 0x7FFFu + ((u >> 16) & 1u);          // round-to-nearest-even
    return (unsigned short)(u >> 16);
}

static __device__ __forceinline__ float leaky(float v) {
    return v > 0.0f ? v : 0.2f * v;
}

// Order-preserving (score, index) packing: float bits made monotonic in the
// high dword, ~index in the low dword so key-ties prefer the smaller index
// (matches lax.top_k tie-breaking). u64 max/min then implement top-2 with
// pure v_cndmask/v_min_u64 selects - no EXEC divergence around the WMMAs.
static __device__ __forceinline__ unsigned long long pack_key(float v, int m) {
    unsigned u = __float_as_uint(v);
    u ^= ((unsigned)((int)u >> 31)) | 0x80000000u;
    return ((unsigned long long)u << 32) | (unsigned)~m;
}
static __device__ __forceinline__ float unpack_val(unsigned long long k) {
    unsigned u = (unsigned)(k >> 32);
    u ^= (u & 0x80000000u) ? 0x80000000u : 0xFFFFFFFFu;
    return __uint_as_float(u);
}
static __device__ __forceinline__ int unpack_idx(unsigned long long k) {
    return (int)~(unsigned)k;
}

#ifdef USE_TDM
// One TDM load of a 16-row x 256-col bf16 tile (global row stride 256 elems)
// into LDS at lds_off; LDS rows padded to 528B (128 DW data + 4 DW pad) so the
// ds_load_b128 fragment reads are bank-conflict free (132 DW lane stride).
static __device__ __forceinline__ void tdm_load_tile(unsigned lds_off,
                                                     const unsigned short* gptr) {
    unsigned long long ga = (unsigned long long)(uintptr_t)gptr;
    u32x4 g0;
    g0[0] = 1u;                                            // count=1, user desc
    g0[1] = lds_off;                                       // lds_addr (bytes)
    g0[2] = (unsigned)(ga & 0xFFFFFFFFu);                  // global_addr[31:0]
    g0[3] = (unsigned)((ga >> 32) & 0x01FFFFFFu) | (2u << 30); // addr[56:32]|type=2

    i32x8 g1;
    // data_size=1(2B) | pad_enable | pad_interval=6(128 DW) | pad_amount=3(4 DW)
    g1[0] = (int)((1u << 16) | (1u << 20) | (6u << 22) | (3u << 25));
    g1[1] = (int)(256u << 16);     // tensor_dim0 = 256      (bits[63:48])
    g1[2] = (int)(512u << 16);     // tensor_dim1 = 512      (bits[95:80])
    g1[3] = (int)(256u << 16);     // tile_dim0   = 256      (bits[127:112])
    g1[4] = 16;                    // tile_dim1   = 16, tile_dim2 = 0
    g1[5] = 256;                   // tensor_dim0_stride = 256 elems
    g1[6] = 0;
    g1[7] = 0;

    i32x4 gz;
    gz[0] = 0; gz[1] = 0; gz[2] = 0; gz[3] = 0;
#if defined(__clang_major__) && (__clang_major__ >= 23)
    i32x8 g4;
    g4[0] = 0; g4[1] = 0; g4[2] = 0; g4[3] = 0;
    g4[4] = 0; g4[5] = 0; g4[6] = 0; g4[7] = 0;
    __builtin_amdgcn_tensor_load_to_lds(g0, g1, gz, gz, g4, 0);
#else
    __builtin_amdgcn_tensor_load_to_lds(g0, g1, gz, gz, 0);
#endif
}
#endif // USE_TDM

// ---------------------------------------------------------------------------
// K1: spatial mean  I_G[b,c] = mean over HW
// ---------------------------------------------------------------------------
__global__ __launch_bounds__(256) void k_mean(const float* __restrict__ img,
                                              float* __restrict__ ig) {
    int bc = blockIdx.x;
    const float* src = img + (size_t)bc * HWN;
    float s = 0.0f;
    for (int i = threadIdx.x; i < HWN; i += 256) s += src[i];
    __shared__ float red[256];
    red[threadIdx.x] = s;
    __syncthreads();
    for (int st = 128; st > 0; st >>= 1) {
        if (threadIdx.x < st) red[threadIdx.x] += red[threadIdx.x + st];
        __syncthreads();
    }
    if (threadIdx.x == 0) ig[bc] = red[0] * (1.0f / (float)HWN);
}

// ---------------------------------------------------------------------------
// K2: global path -> gate[b,c] = sigmoid( softmax(I_G @ mem^T) @ mem + I_G )
// ---------------------------------------------------------------------------
__global__ __launch_bounds__(256) void k_global(const float* __restrict__ ig,
                                                const float* __restrict__ mem,
                                                float* __restrict__ gate) {
    int b = blockIdx.x, t = threadIdx.x;
    __shared__ float sIG[CN];
    __shared__ float sSc[MN];
    __shared__ float sRed[256];

    sIG[t] = ig[b * CN + t];
    __syncthreads();

    for (int m = t; m < MN; m += 256) {
        const float* mr = mem + (size_t)m * CN;
        float s = 0.0f;
        for (int c = 0; c < CN; ++c) s = fmaf(sIG[c], mr[c], s);
        sSc[m] = s;
    }
    __syncthreads();

    float mx = fmaxf(sSc[t], sSc[t + 256]);
    sRed[t] = mx; __syncthreads();
    for (int st = 128; st > 0; st >>= 1) {
        if (t < st) sRed[t] = fmaxf(sRed[t], sRed[t + st]);
        __syncthreads();
    }
    mx = sRed[0];
    __syncthreads();
    float e0 = __expf(sSc[t] - mx), e1 = __expf(sSc[t + 256] - mx);
    sSc[t] = e0; sSc[t + 256] = e1;
    sRed[t] = e0 + e1; __syncthreads();
    for (int st = 128; st > 0; st >>= 1) {
        if (t < st) sRed[t] += sRed[t + st];
        __syncthreads();
    }
    float inv = 1.0f / sRed[0];
    __syncthreads();

    float r = 0.0f;
    for (int m = 0; m < MN; ++m) r = fmaf(sSc[m], mem[(size_t)m * CN + t], r);
    r = r * inv + sIG[t];
    gate[b * CN + t] = 1.0f / (1.0f + __expf(-r));
}

// ---------------------------------------------------------------------------
// K3: convert memory (M*C) and fusion_w (C*2C) to bf16
// ---------------------------------------------------------------------------
__global__ __launch_bounds__(256) void k_cvt(const float* __restrict__ mem,
                                             const float* __restrict__ fw,
                                             unsigned short* __restrict__ memb,
                                             unsigned short* __restrict__ fwb) {
    int i = blockIdx.x * 256 + threadIdx.x;
    if (i < MN * CN) {
        memb[i] = f2bf(mem[i]);
        fwb[i]  = f2bf(fw[i]);
    }
}

// ---------------------------------------------------------------------------
// K4: tiled transpose [b][c][p] -> channel-contiguous bf16 rows
// ---------------------------------------------------------------------------
__global__ __launch_bounds__(256) void k_tg(const float* __restrict__ img,
                                            const float* __restrict__ gate,
                                            unsigned short* __restrict__ xraw,
                                            unsigned short* __restrict__ fusedA) {
    int b  = blockIdx.z;
    int c0 = blockIdx.y * 64;
    int p0 = blockIdx.x * 64;
    int t  = threadIdx.x;
    __shared__ float tile[64][65];

    const float* src = img + ((size_t)b * CN + c0) * HWN + p0;
    for (int it = 0; it < 16; ++it) {
        int cl = it * 4 + (t >> 6);
        int pl = t & 63;
        tile[pl][cl] = src[(size_t)cl * HWN + pl];
    }
    __syncthreads();
    for (int it = 0; it < 16; ++it) {
        int pl = it * 4 + (t >> 6);
        int cl = t & 63;
        float v = tile[pl][cl];
        size_t row = (size_t)b * HWN + p0 + pl;
        xraw[row * CN + c0 + cl] = f2bf(v);
        float g = gate[b * CN + c0 + cl];
        fusedA[row * (2 * CN) + c0 + cl] = f2bf(v * g);
    }
}

// ---------------------------------------------------------------------------
// K5: score GEMM (bf16 WMMA) fused with per-pixel top-2 + softmax + blend.
// B tiles staged by the Tensor Data Mover: triple-buffered LDS, two DMAs in
// flight, one workgroup barrier per tile. Top-2 kept as packed u64 keys so
// the whole reduction is select-only (EXEC stays all-ones around WMMA).
// ---------------------------------------------------------------------------
#define BROW 528   // padded LDS row stride in bytes (512 data + 16 pad)

__global__ __launch_bounds__(128) void k_score_topk(
        const unsigned short* __restrict__ Xraw,
        const unsigned short* __restrict__ memb,
        const float* __restrict__ memf,
        unsigned short* __restrict__ fusedA) {
    int blk  = blockIdx.x;                  // B * (HW/64) = 512
    int b    = blk >> 6;
    int p0   = (blk & 63) * 64;
    int t    = threadIdx.x;
    int wave = t >> 5;
    int lane = t & 31;
    int h    = lane >> 4;                   // lane half
    int ln   = lane & 15;
    int pw   = p0 + wave * 16;              // this wave's pixel base

#ifdef USE_TDM
    __shared__ __align__(16) unsigned char sB[3][16 * BROW];
#endif
    __shared__ int   sI0[64], sI1[64];
    __shared__ float sA0[64];

    // A fragments resident in VGPRs: lane row = pixel pw+ln,
    // K-step kk: lanes 0-15 hold K=kk*32+[0..7],[16..23]; lanes 16-31 +8.
    Frag A[8];
    const unsigned short* arow = Xraw + ((size_t)b * HWN + pw + ln) * CN;
    #pragma unroll
    for (int k = 0; k < 8; ++k) {
        int base = k * 32 + h * 8;
        A[k].h[0] = *(const bf16x8*)(arow + base);
        A[k].h[1] = *(const bf16x8*)(arow + base + 16);
    }

    unsigned long long k0[8], k1[8];
    #pragma unroll
    for (int s = 0; s < 8; ++s) { k0[s] = 0ull; k1[s] = 0ull; }

#ifdef USE_TDM
    unsigned l0 = (unsigned)(uintptr_t)&sB[0][0];
    unsigned l1 = (unsigned)(uintptr_t)&sB[1][0];
    unsigned l2 = (unsigned)(uintptr_t)&sB[2][0];
    if (wave == 0) {
        tdm_load_tile(l0, memb);                       // tiles 0 and 1 in flight
        tdm_load_tile(l1, memb + 16 * CN);
    }
#endif

    for (int mt = 0; mt < 32; ++mt) {
#ifdef USE_TDM
        if (wave == 0) {
            if (mt < 31) __builtin_amdgcn_s_wait_tensorcnt(1);  // tile mt landed
            else         __builtin_amdgcn_s_wait_tensorcnt(0);
        }
        __syncthreads();     // tile mt visible; everyone done reading tile mt-1
        if (wave == 0 && mt < 30) {
            int nb = (mt + 2) % 3;                     // overwrites tile mt-1 buf
            unsigned ld = (nb == 0) ? l0 : ((nb == 1) ? l1 : l2);
            tdm_load_tile(ld, memb + (size_t)(mt + 2) * 16 * CN);
        }
        const unsigned char* bt =
            &sB[0][0] + (size_t)(mt % 3) * (16 * BROW) + (size_t)ln * BROW;
#else
        const unsigned short* brow = memb + (size_t)(mt * 16 + ln) * CN;
#endif
        Frag Bf[8];
        #pragma unroll
        for (int k = 0; k < 8; ++k) {
#ifdef USE_TDM
            int boff = k * 64 + h * 16;
            Bf[k].h[0] = *(const bf16x8*)(bt + boff);
            Bf[k].h[1] = *(const bf16x8*)(bt + boff + 32);
#else
            int base = k * 32 + h * 8;
            Bf[k].h[0] = *(const bf16x8*)(brow + base);
            Bf[k].h[1] = *(const bf16x8*)(brow + base + 16);
#endif
        }
        sched_fence();       // issue all 16 ds_load_b128 before the WMMA chain
        v8f acc;
        #pragma unroll
        for (int i = 0; i < 8; ++i) acc[i] = 0.0f;
        #pragma unroll
        for (int k = 0; k < 8; ++k) {
            acc = __builtin_amdgcn_wmma_f32_16x16x32_bf16(
                      false, A[k].f, false, Bf[k].f, (short)0, acc, false, false);
        }
        int mval = mt * 16 + ln;
        #pragma unroll
        for (int s = 0; s < 8; ++s) {                  // branch-free top-2 update
            unsigned long long kk = pack_key(acc[s], mval);
            unsigned long long a  = k0[s];
            unsigned long long hi = (kk > a) ? kk : a;
            unsigned long long lo = (kk > a) ? a  : kk;
            k0[s] = hi;
            k1[s] = (lo > k1[s]) ? lo : k1[s];
        }
    }

    // branch-free top-2 merge across the 16 lanes of each half:
    // top = max(a0,b0); second = max(min(a0,b0), max(a1,b1))  (pairs sorted)
    #pragma unroll
    for (int mask = 1; mask <= 8; mask <<= 1) {
        #pragma unroll
        for (int s = 0; s < 8; ++s) {
            unsigned long long o0 =
                (unsigned long long)__shfl_xor((long long)k0[s], mask);
            unsigned long long o1 =
                (unsigned long long)__shfl_xor((long long)k1[s], mask);
            unsigned long long a  = k0[s];
            unsigned long long t0 = (a > o0) ? a  : o0;
            unsigned long long tm = (a > o0) ? o0 : a;
            unsigned long long s1 = (k1[s] > o1) ? k1[s] : o1;
            k0[s] = t0;
            k1[s] = (tm > s1) ? tm : s1;
        }
    }

    if (ln == 0) {
        #pragma unroll
        for (int s = 0; s < 8; ++s) {
            int pl = wave * 16 + h * 8 + s;
            float v0 = unpack_val(k0[s]);
            float v1 = unpack_val(k1[s]);
            sI0[pl] = unpack_idx(k0[s]);
            sI1[pl] = unpack_idx(k1[s]);
            sA0[pl] = 1.0f / (1.0f + __expf(v1 - v0));
        }
    }
    __syncthreads();

    // mem_feat[p,c] = a0*mem[i0,c] + a1*mem[i1,c]  -> fusedA[...,256:512]
    int c2 = t * 2;
    for (int p = 0; p < 64; ++p) {
        int   i0 = sI0[p], i1 = sI1[p];
        float a0 = sA0[p], a1 = 1.0f - a0;
        const float* r0 = memf + (size_t)i0 * CN + c2;
        const float* r1 = memf + (size_t)i1 * CN + c2;
        unsigned lo = f2bf(a0 * r0[0] + a1 * r1[0]);
        unsigned hi = f2bf(a0 * r0[1] + a1 * r1[1]);
        size_t ro = ((size_t)b * HWN + p0 + p) * (2 * CN) + CN + c2;
        *(unsigned*)(fusedA + ro) = lo | (hi << 16);
    }
}

// ---------------------------------------------------------------------------
// K6: fusion GEMM (bf16 WMMA): x[b,o,p] = leaky( fusedA[p,:]·fusion_w[o,:]+b[o] )
// ---------------------------------------------------------------------------
__global__ __launch_bounds__(128) void k_fusion(
        const unsigned short* __restrict__ fusedA,
        const unsigned short* __restrict__ fwb,
        const float* __restrict__ fb,
        float* __restrict__ xbuf) {
    int blk  = blockIdx.x;                  // 512
    int b    = blk >> 6;
    int p0   = (blk & 63) * 64;
    int t    = threadIdx.x;
    int wave = t >> 5;
    int lane = t & 31;
    int h    = lane >> 4;
    int ln   = lane & 15;
    int pw   = p0 + wave * 16;

    const unsigned short* arow = fusedA + ((size_t)b * HWN + pw + ln) * (2 * CN);

    for (int pass = 0; pass < 2; ++pass) {
        v8f acc[8];
        #pragma unroll
        for (int ot = 0; ot < 8; ++ot)
            #pragma unroll
            for (int i = 0; i < 8; ++i) acc[ot][i] = 0.0f;

        for (int k = 0; k < 16; ++k) {      // K = 512 / 32
            int base = k * 32 + h * 8;
            Frag Af;
            Af.h[0] = *(const bf16x8*)(arow + base);
            Af.h[1] = *(const bf16x8*)(arow + base + 16);
            Frag Bf[8];
            #pragma unroll
            for (int ot = 0; ot < 8; ++ot) {
                int o = (pass * 8 + ot) * 16 + ln;
                const unsigned short* brow = fwb + (size_t)o * (2 * CN);
                Bf[ot].h[0] = *(const bf16x8*)(brow + base);
                Bf[ot].h[1] = *(const bf16x8*)(brow + base + 16);
            }
            sched_fence();   // batch the 18 global_load_b128 ahead of the WMMAs
            #pragma unroll
            for (int ot = 0; ot < 8; ++ot) {
                acc[ot] = __builtin_amdgcn_wmma_f32_16x16x32_bf16(
                              false, Af.f, false, Bf[ot].f, (short)0, acc[ot], false, false);
            }
        }
        #pragma unroll
        for (int ot = 0; ot < 8; ++ot) {
            int o = (pass * 8 + ot) * 16 + ln;
            float bias = fb[o];
            float* dst = xbuf + ((size_t)b * CN + o) * HWN + pw + h * 8;
            #pragma unroll
            for (int s = 0; s < 8; ++s) dst[s] = leaky(acc[ot][s] + bias);
        }
    }
}

// ---------------------------------------------------------------------------
// K7: depthwise dilated 3x3 (dilation 2, pad 2) + bias + leaky
// ---------------------------------------------------------------------------
__global__ __launch_bounds__(256) void k_depthwise(
        const float* __restrict__ x,
        const float* __restrict__ dw,
        const float* __restrict__ db,
        float* __restrict__ out) {
    size_t idx = (size_t)blockIdx.x * 256 + threadIdx.x;
    int w  = (int)(idx & 63);
    int hh = (int)((idx >> 6) & 63);
    int c  = (int)((idx >> 12) & (CN - 1));
    const float* xc = x + (idx & ~(size_t)(HWN - 1));
    const float* wf = dw + c * 9;
    float acc = db[c];
    #pragma unroll
    for (int i = 0; i < 3; ++i) {
        int yy = hh + (i - 1) * 2;
        if (yy < 0 || yy > 63) continue;
        #pragma unroll
        for (int j = 0; j < 3; ++j) {
            int xx = w + (j - 1) * 2;
            if (xx < 0 || xx > 63) continue;
            acc = fmaf(wf[i * 3 + j], xc[yy * WN + xx], acc);
        }
    }
    out[idx] = leaky(acc);
}

// ---------------------------------------------------------------------------
// Workspace layout (bytes, all 4KB-aligned)
// ---------------------------------------------------------------------------
#define IG_OFF      ((size_t)0)                       //  8 KB : I_G
#define GATE_OFF    ((size_t)8192)                    //  8 KB : gate
#define MEMB_OFF    ((size_t)16384)                   // 256 KB: memory bf16
#define FWB_OFF     ((size_t)278528)                  // 256 KB: fusion_w bf16
#define XRAW_OFF    ((size_t)540672)                  //  16 MB: img transposed bf16
#define FUSEDA_OFF  ((size_t)17317888)                //  32 MB: fusion-A bf16
#define XBUF_OFF    ((size_t)50872320)                //  32 MB: fused conv out f32

extern "C" void kernel_launch(void* const* d_in, const int* in_sizes, int n_in,
                              void* d_out, int out_size, void* d_ws, size_t ws_size,
                              hipStream_t stream) {
    const float* img = (const float*)d_in[0];   // [8,256,64,64]
    const float* mem = (const float*)d_in[1];   // [512,256]
    const float* fw  = (const float*)d_in[2];   // [256,512]
    const float* fb  = (const float*)d_in[3];   // [256]
    const float* dwp = (const float*)d_in[4];   // [256,1,3,3]
    const float* dbp = (const float*)d_in[5];   // [256]
    float* out = (float*)d_out;

    char* ws = (char*)d_ws;
    float*          IG     = (float*)(ws + IG_OFF);
    float*          GATE   = (float*)(ws + GATE_OFF);
    unsigned short* MEMB   = (unsigned short*)(ws + MEMB_OFF);
    unsigned short* FWB    = (unsigned short*)(ws + FWB_OFF);
    unsigned short* XRAW   = (unsigned short*)(ws + XRAW_OFF);
    unsigned short* FUSEDA = (unsigned short*)(ws + FUSEDA_OFF);
    float*          XBUF   = (float*)(ws + XBUF_OFF);

    k_mean      <<<BN * CN, 256, 0, stream>>>(img, IG);
    k_global    <<<BN, 256, 0, stream>>>(IG, mem, GATE);
    k_cvt       <<<(MN * CN) / 256, 256, 0, stream>>>(mem, fw, MEMB, FWB);
    k_tg        <<<dim3(HWN / 64, CN / 64, BN), 256, 0, stream>>>(img, GATE, XRAW, FUSEDA);
    k_score_topk<<<BN * (HWN / 64), 128, 0, stream>>>(XRAW, MEMB, mem, FUSEDA);
    k_fusion    <<<BN * (HWN / 64), 128, 0, stream>>>(FUSEDA, FWB, fb, XBUF);
    k_depthwise <<<(BN * CN * HWN) / 256, 256, 0, stream>>>(XBUF, dwp, dbp, out);
}
